// ATM_24635932410067
// MI455X (gfx1250) — matrix operand order repaired and checked by reading, not compiled, not run on hardware
//
#include <hip/hip_runtime.h>
#include <hip/hip_bf16.h>

// Problem constants (from reference)
#define BATCH 2
#define NQS   2048
#define NKS   4096
#define DIMC  256
#define NH    8
#define HDIM  32
#define SCALE 0.17677669529663687f   // 32^-0.5

typedef _Float16 v16h  __attribute__((ext_vector_type(16)));
typedef float    v8f   __attribute__((ext_vector_type(8)));
typedef _Float16 f16x2 __attribute__((ext_vector_type(2)));

static __device__ __forceinline__ v8f wmma_f16(v16h a, v16h b, v8f c) {
  // emits v_wmma_f32_16x16x32_f16
  return __builtin_amdgcn_wmma_f32_16x16x32_f16(false, a, false, b, (short)0, c,
                                                false, false);
}

// A-matrix (16x32 f16) per-lane K base for packed pair j (CDNA5 ISA 7.12.2)
static __device__ __forceinline__ int a_k0(int j, int half) {
  return ((j >= 4) ? 16 : 0) + 2 * (j & 3) + 8 * half;
}
// B-matrix (32x16 f16) per-lane K base for packed pair j
static __device__ __forceinline__ int b_k0(int j, int half) {
  return ((j >= 4) ? 16 : 0) + 2 * (j & 3) + 8 * half;
}

// ds_swizzle_b32 xor-butterfly within 32 lanes: offset15=0, and=0x1F, xor=MASK
template <int MASK>
static __device__ __forceinline__ float swz_xor(float v) {
  return __int_as_float(
      __builtin_amdgcn_ds_swizzle(__float_as_int(v), (MASK << 10) | 0x1F));
}
// max/sum reduction across the 16-lane half-wave (masks < 16 stay in-half,
// matching the WMMA C/D row split: lanes 0-15 = rows 0-7, 16-31 = rows 8-15)
static __device__ __forceinline__ float red_max16(float v) {
  v = fmaxf(v, swz_xor<1>(v));
  v = fmaxf(v, swz_xor<2>(v));
  v = fmaxf(v, swz_xor<4>(v));
  v = fmaxf(v, swz_xor<8>(v));
  return v;
}
static __device__ __forceinline__ float red_sum16(float v) {
  v += swz_xor<1>(v);
  v += swz_xor<2>(v);
  v += swz_xor<4>(v);
  v += swz_xor<8>(v);
  return v;
}

// ---- CDNA5 async global->LDS copies (ASYNCcnt-tracked, no VGPR round-trip).
// GV mode: 64-bit vaddr, no saddr; INST_OFFSET applies to both LDS and global
// addresses (ISA 10.x async pseudocode), so one base pair serves 4 x b128.
template <int OFF>
static __device__ __forceinline__ void async_ld_b128(unsigned lds, const void* g) {
  asm volatile("global_load_async_to_lds_b128 %0, %1, off offset:%2"
               :: "v"(lds), "v"(g), "i"(OFF) : "memory");
}
static __device__ __forceinline__ void async_ld_b32(unsigned lds, const void* g) {
  asm volatile("global_load_async_to_lds_b32 %0, %1, off"
               :: "v"(lds), "v"(g) : "memory");
}
static __device__ __forceinline__ void wait_async0() {
  asm volatile("s_wait_asynccnt 0" ::: "memory");
}
static __device__ __forceinline__ unsigned lds_addr(const void* p) {
  return (unsigned)(unsigned long long)p;   // low 32 bits = LDS aperture offset
}

// ---------------------------------------------------------------------------
// Kernel 1: projection  Y[row,d] = sum_c X[row,c] * W[d,c], output f16.
// TRANSPOSED=0: out[((b*NH+h)*N + n)*HD + hd]   (Q, K layout)
// TRANSPOSED=1: out[((b*NH+h)*HD + hd)*N + n]   (V transposed layout)
// Block: 256 threads = 8 waves; 16-row strip x 256 cols; wave w owns d-tiles
// 2w, 2w+1; 8 K-steps of 32.  N = 1<<NSHIFT, compile-time.
// ---------------------------------------------------------------------------
template <int NSHIFT, bool TRANSPOSED>
__global__ __launch_bounds__(256) void proj_kernel(
    const float* __restrict__ X, const float* __restrict__ W,
    _Float16* __restrict__ out) {
  __shared__ __align__(16) _Float16 Xl[16][32];
  __shared__ __align__(16) _Float16 Wl[256][32];

  const int tid  = threadIdx.x;
  const int lane = tid & 31;
  const int wv   = tid >> 5;
  const int half = lane >> 4;
  const int ln   = lane & 15;

  v8f acc[2] = {v8f{}, v8f{}};

  for (int cs = 0; cs < DIMC; cs += 32) {
    __syncthreads();
    // stage X strip tile [16][32], f32 -> f16 (float2 per thread)
    {
      int m = tid >> 4, c = (tid & 15) * 2;
      float2 v = *(const float2*)(X + (blockIdx.x * 16 + m) * DIMC + cs + c);
      Xl[m][c]     = (_Float16)v.x;
      Xl[m][c + 1] = (_Float16)v.y;
    }
    // stage W slab [256 d][32 c]: thread t loads row d=t
    {
      const float4* wr = (const float4*)(W + tid * DIMC + cs);
#pragma unroll
      for (int q = 0; q < 8; ++q) {
        float4 v = wr[q];
        Wl[tid][q * 4 + 0] = (_Float16)v.x;
        Wl[tid][q * 4 + 1] = (_Float16)v.y;
        Wl[tid][q * 4 + 2] = (_Float16)v.z;
        Wl[tid][q * 4 + 3] = (_Float16)v.w;
      }
    }
    __syncthreads();

    v16h a;
#pragma unroll
    for (int j = 0; j < 8; ++j) {
      f16x2 t = *(const f16x2*)&Xl[ln][a_k0(j, half)];
      a[2 * j] = t[0]; a[2 * j + 1] = t[1];
    }
#pragma unroll
    for (int t2 = 0; t2 < 2; ++t2) {
      v16h bm;
#pragma unroll
      for (int j = 0; j < 8; ++j) {
        f16x2 t = *(const f16x2*)&Wl[(2 * wv + t2) * 16 + ln][b_k0(j, half)];
        bm[2 * j] = t[0]; bm[2 * j + 1] = t[1];
      }
      acc[t2] = wmma_f16(a, bm, acc[t2]);
    }
  }

  constexpr int N = 1 << NSHIFT;
#pragma unroll
  for (int t2 = 0; t2 < 2; ++t2) {
    const int d  = (2 * wv + t2) * 16 + ln;
    const int hh = d >> 5, hd = d & 31;
#pragma unroll
    for (int r = 0; r < 8; ++r) {
      int m   = r + 8 * half;
      int row = blockIdx.x * 16 + m;
      int bi  = row >> NSHIFT;
      int n   = row & (N - 1);
      long idx = TRANSPOSED
                     ? (((long)(bi * NH + hh) * HDIM + hd) * N + n)
                     : (((long)(bi * NH + hh) * N + n) * HDIM + hd);
      out[idx] = (_Float16)acc[t2][r];
    }
  }
}

// ---------------------------------------------------------------------------
// Kernel 2: flash attention per (batch, 16-query strip). 8 waves = 8 heads.
// Per 32-wide key chunk: 2 score WMMAs + online softmax + 2 PV WMMAs per head.
// K/V/Q tiles staged with global_load_async_to_lds_b128 (ASYNCcnt).
// Dynamic LDS layout (64 KB):
//   Sl [8][16][32] f32 (16K) | Kl [8][32][32] f16 (16K) |
//   Vl [8][32][32] f16 (16K) | Ql [8][16][32] f16 (8K)  | Pl [8][16][32] f16 (8K)
// ---------------------------------------------------------------------------
__global__ __launch_bounds__(256) void attn_kernel(
    const _Float16* __restrict__ Qg, const _Float16* __restrict__ Kg,
    const _Float16* __restrict__ Vtg, _Float16* __restrict__ Xg,
    float* __restrict__ attn_out) {
  extern __shared__ __align__(16) char smem[];
  float (*Sl)[16][32]    = (float (*)[16][32])(smem);
  _Float16 (*Kl)[32][32] = (_Float16 (*)[32][32])(smem + 16384);
  _Float16 (*Vl)[32][32] = (_Float16 (*)[32][32])(smem + 32768);
  _Float16 (*Ql)[16][32] = (_Float16 (*)[16][32])(smem + 49152);
  _Float16 (*Pl)[16][32] = (_Float16 (*)[16][32])(smem + 57344);

  const int tid  = threadIdx.x;
  const int lane = tid & 31;
  const int h    = tid >> 5;    // wave id == head id
  const int half = lane >> 4;
  const int ln   = lane & 15;
  const int bi   = blockIdx.x >> 7;        // batch
  const int q0   = (blockIdx.x & 127) * 16;

  // ---- stage Q strip for all heads: [8][16][32] f16 (async to LDS) ----
  {
    int sh  = tid >> 5;
    int rem = tid & 31;
    int m   = rem >> 1;
    int hd0 = (rem & 1) * 16;
    const void* gq =
        (const void*)(Qg + ((long)(bi * NH + sh) * NQS + q0 + m) * HDIM + hd0);
    unsigned lq = lds_addr(&Ql[sh][m][hd0]);
    async_ld_b128<0>(lq, gq);
    async_ld_b128<16>(lq, gq);
  }
  wait_async0();
  __syncthreads();

  v16h Qa;
#pragma unroll
  for (int j = 0; j < 8; ++j) {
    f16x2 t = *(const f16x2*)&Ql[h][ln][a_k0(j, half)];
    Qa[2 * j] = t[0]; Qa[2 * j + 1] = t[1];
  }

  v8f O0 = {}, O1 = {};
  float rowmax[8], rowsum[8];
#pragma unroll
  for (int r = 0; r < 8; ++r) { rowmax[r] = -1.0e30f; rowsum[r] = 0.0f; }

  for (int kc = 0; kc < NKS; kc += 32) {
    __syncthreads();
    // ---- stage K chunk [8][32 kn][32 hd] + V^T chunk [8][32 hd][32 kn] ----
    {
      int sh = tid >> 5, i = tid & 31;
      const void* gk =
          (const void*)(Kg + ((long)(bi * NH + sh) * NKS + kc + i) * HDIM);
      unsigned lk = lds_addr(&Kl[sh][i][0]);
      async_ld_b128<0>(lk, gk);
      async_ld_b128<16>(lk, gk);
      async_ld_b128<32>(lk, gk);
      async_ld_b128<48>(lk, gk);
      const void* gv =
          (const void*)(Vtg + ((long)(bi * NH + sh) * HDIM + i) * NKS + kc);
      unsigned lv = lds_addr(&Vl[sh][i][0]);
      async_ld_b128<0>(lv, gv);
      async_ld_b128<16>(lv, gv);
      async_ld_b128<32>(lv, gv);
      async_ld_b128<48>(lv, gv);
      if (kc + 32 < NKS) {                  // global_prefetch_b8 of next chunk
        __builtin_prefetch((const char*)gk + 32 * HDIM * 2, 0, 1);
        __builtin_prefetch((const char*)gv + 64, 0, 1);
      }
    }
    wait_async0();
    __syncthreads();

    // ---- scores: S[16q x 32k] = Q(16x32) @ K^T, two 16x16 WMMA tiles ----
    v16h kb0, kb1;
#pragma unroll
    for (int j = 0; j < 8; ++j) {
      int k0 = b_k0(j, half);
      f16x2 t0 = *(const f16x2*)&Kl[h][ln][k0];
      f16x2 t1 = *(const f16x2*)&Kl[h][16 + ln][k0];
      kb0[2 * j] = t0[0]; kb0[2 * j + 1] = t0[1];
      kb1[2 * j] = t1[0]; kb1[2 * j + 1] = t1[1];
    }
    v8f S0 = wmma_f16(Qa, kb0, v8f{});
    v8f S1 = wmma_f16(Qa, kb1, v8f{});
#pragma unroll
    for (int r = 0; r < 8; ++r) {
      S0[r] *= SCALE; S1[r] *= SCALE;
      int m = r + 8 * half;
      Sl[h][m][ln]      = S0[r];
      Sl[h][m][16 + ln] = S1[r];
    }
    __syncthreads();

    // ---- attn_save: head-mean of scaled scores (one float2 per thread) ----
    {
      int m = tid >> 4, n = (tid & 15) * 2;
      float s0 = 0.0f, s1 = 0.0f;
#pragma unroll
      for (int hh = 0; hh < NH; ++hh) {
        s0 += Sl[hh][m][n];
        s1 += Sl[hh][m][n + 1];
      }
      float2 st; st.x = s0 * 0.125f; st.y = s1 * 0.125f;
      *(float2*)(attn_out + ((long)(bi * NQS + q0 + m)) * NKS + kc + n) = st;
    }

    // ---- online softmax (per wave/head) ----
#pragma unroll
    for (int r = 0; r < 8; ++r) {
      float v  = red_max16(fmaxf(S0[r], S1[r]));  // row max (16-lane half)
      float nm = fmaxf(rowmax[r], v);
      float f  = __expf(rowmax[r] - nm);
      rowmax[r] = nm;
      float p0 = __expf(S0[r] - nm);
      float p1 = __expf(S1[r] - nm);
      rowsum[r] = rowsum[r] * f + p0 + p1;  // per-lane partial (own columns)
      O0[r] *= f; O1[r] *= f;
      int m = r + 8 * half;
      Pl[h][m][ln]      = (_Float16)p0;     // stage P for A-layout reload
      Pl[h][m][16 + ln] = (_Float16)p1;
    }
    __syncthreads();

    // ---- PV: O(16x32) += P(16x32) @ V(32x32), two 16x16 WMMA tiles ----
    v16h pa, vb0, vb1;
#pragma unroll
    for (int j = 0; j < 8; ++j) {
      f16x2 tp = *(const f16x2*)&Pl[h][ln][a_k0(j, half)];
      pa[2 * j] = tp[0]; pa[2 * j + 1] = tp[1];
      int k0 = b_k0(j, half);
      f16x2 t0 = *(const f16x2*)&Vl[h][ln][k0];
      f16x2 t1 = *(const f16x2*)&Vl[h][16 + ln][k0];
      vb0[2 * j] = t0[0]; vb0[2 * j + 1] = t0[1];
      vb1[2 * j] = t1[0]; vb1[2 * j + 1] = t1[1];
    }
    O0 = wmma_f16(pa, vb0, O0);
    O1 = wmma_f16(pa, vb1, O1);
  }

  // ---- finalize: divide by row sums, write X[B,NQ,H*HD] f16 ----
#pragma unroll
  for (int r = 0; r < 8; ++r) {
    float s   = red_sum16(rowsum[r]);
    float inv = 1.0f / s;
    int m = r + 8 * half;
    long row = (long)(bi * NQS + q0 + m);
    Xg[row * DIMC + h * HDIM + ln]      = (_Float16)(O0[r] * inv);
    Xg[row * DIMC + h * HDIM + 16 + ln] = (_Float16)(O1[r] * inv);
  }
}

// ---------------------------------------------------------------------------
// Kernel 3: output projection  out[row,d] = sum_c X[row,c]*Wp[d,c] + bp[d], f32
// X tile staged async (b32 per thread), W slab converted through VGPRs.
// ---------------------------------------------------------------------------
__global__ __launch_bounds__(256) void oproj_kernel(
    const _Float16* __restrict__ Xg, const float* __restrict__ W,
    const float* __restrict__ bias, float* __restrict__ out) {
  __shared__ __align__(16) _Float16 Xl[16][32];
  __shared__ __align__(16) _Float16 Wl[256][32];

  const int tid  = threadIdx.x;
  const int lane = tid & 31;
  const int wv   = tid >> 5;
  const int half = lane >> 4;
  const int ln   = lane & 15;

  v8f acc[2] = {v8f{}, v8f{}};

  for (int cs = 0; cs < DIMC; cs += 32) {
    __syncthreads();
    {
      int m = tid >> 4, c = (tid & 15) * 2;
      async_ld_b32(lds_addr(&Xl[m][c]),
                   (const void*)(Xg + (blockIdx.x * 16 + m) * DIMC + cs + c));
    }
    {
      const float4* wr = (const float4*)(W + tid * DIMC + cs);
#pragma unroll
      for (int q = 0; q < 8; ++q) {
        float4 v = wr[q];
        Wl[tid][q * 4 + 0] = (_Float16)v.x;
        Wl[tid][q * 4 + 1] = (_Float16)v.y;
        Wl[tid][q * 4 + 2] = (_Float16)v.z;
        Wl[tid][q * 4 + 3] = (_Float16)v.w;
      }
    }
    wait_async0();
    __syncthreads();

    v16h a;
#pragma unroll
    for (int j = 0; j < 8; ++j) {
      f16x2 t = *(const f16x2*)&Xl[ln][a_k0(j, half)];
      a[2 * j] = t[0]; a[2 * j + 1] = t[1];
    }
#pragma unroll
    for (int t2 = 0; t2 < 2; ++t2) {
      v16h bm;
#pragma unroll
      for (int j = 0; j < 8; ++j) {
        f16x2 t = *(const f16x2*)&Wl[(2 * wv + t2) * 16 + ln][b_k0(j, half)];
        bm[2 * j] = t[0]; bm[2 * j + 1] = t[1];
      }
      acc[t2] = wmma_f16(a, bm, acc[t2]);
    }
  }

#pragma unroll
  for (int t2 = 0; t2 < 2; ++t2) {
    int d = (2 * wv + t2) * 16 + ln;
    float bb = bias[d];
#pragma unroll
    for (int r = 0; r < 8; ++r) {
      int m = r + 8 * half;
      out[(long)(blockIdx.x * 16 + m) * DIMC + d] = acc[t2][r] + bb;
    }
  }
}

// ---------------------------------------------------------------------------
extern "C" void kernel_launch(void* const* d_in, const int* in_sizes, int n_in,
                              void* d_out, int out_size, void* d_ws,
                              size_t ws_size, hipStream_t stream) {
  (void)in_sizes; (void)n_in; (void)out_size; (void)ws_size;
  const float* xq = (const float*)d_in[0];
  const float* xk = (const float*)d_in[1];
  const float* xv = (const float*)d_in[2];
  const float* wq = (const float*)d_in[3];
  const float* wk = (const float*)d_in[4];
  const float* wv = (const float*)d_in[5];
  const float* wp = (const float*)d_in[6];
  const float* bp = (const float*)d_in[7];

  float* out      = (float*)d_out;
  float* attn_out = out + (long)BATCH * NQS * DIMC;   // second output tensor

  _Float16* Qh  = (_Float16*)d_ws;                          // 2 MB
  _Float16* Kh  = Qh  + (long)BATCH * NH * NQS * HDIM;      // 4 MB
  _Float16* Vth = Kh  + (long)BATCH * NH * NKS * HDIM;      // 4 MB
  _Float16* Xh  = Vth + (long)BATCH * NH * NKS * HDIM;      // 2 MB

  proj_kernel<11, false><<<(BATCH * NQS) / 16, 256, 0, stream>>>(xq, wq, Qh);
  proj_kernel<12, false><<<(BATCH * NKS) / 16, 256, 0, stream>>>(xk, wk, Kh);
  proj_kernel<12, true><<<(BATCH * NKS) / 16, 256, 0, stream>>>(xv, wv, Vth);
  attn_kernel<<<BATCH * (NQS / 16), 256, 65536, stream>>>(Qh, Kh, Vth, Xh,
                                                          attn_out);
  oproj_kernel<<<(BATCH * NQS) / 16, 256, 0, stream>>>(Xh, wp, bp, out);
}
